// causal_multihead_self_attention_34196529610911
// MI455X (gfx1250) — matrix operand order repaired and checked by reading, not compiled, not run on hardware
//
#include <hip/hip_runtime.h>
#include <hip/hip_bf16.h>

// MI455X (gfx1250) causal MHA with RoPE.
// Pipeline: fp32->bf16 convert -> QKV GEMM (bf16 WMMA, fused RoPE) ->
// flash attention (bf16 WMMA, online softmax) -> output proj GEMM (bf16 WMMA).
// GEMM tiles staged with GLOBAL_LOAD_ASYNC_TO_LDS_B128 (double-buffered).
// Workspace layout (requires >= 88 MB):
//   [0,16M)   x bf16            8192x1024
//   [16,22M)  W_qkv bf16        3072x1024
//   [22,24M)  W_o bf16          1024x1024
//   [24,40M)  Q bf16  [B,H,S,dk]
//   [40,56M)  K bf16  [B,H,S,dk]
//   [56,72M)  V bf16  [B,H,S,dk]
//   [72,88M)  attn out bf16 [B,S,1024]

typedef __bf16 bf16_t;
typedef __attribute__((ext_vector_type(16))) __bf16 v16bf;
typedef __attribute__((ext_vector_type(8)))  __bf16 v8bf;
typedef __attribute__((ext_vector_type(4)))  __bf16 v4bf;
typedef __attribute__((ext_vector_type(8)))  float  v8f;
typedef __attribute__((ext_vector_type(4)))  float  v4f;
typedef __attribute__((ext_vector_type(4)))  int    v4i;

#define S_LEN 2048
#define DMODEL 1024
#define NHEAD 16
#define DK 64

// ---- async global->LDS copy (CDNA5), with synchronous fallback ----
#if defined(__gfx1250__) &&                                           \
    __has_builtin(__builtin_amdgcn_global_load_async_to_lds_b128) &&  \
    __has_builtin(__builtin_amdgcn_s_wait_asynccnt)
#define ASYNC_LDS 1
#endif

// Copy 16 bytes (8 bf16) from global to LDS.
__device__ inline void cp16(bf16_t* l, const bf16_t* g) {
#ifdef ASYNC_LDS
  __builtin_amdgcn_global_load_async_to_lds_b128(
      (__attribute__((address_space(1))) v4i*)g,
      (__attribute__((address_space(3))) v4i*)l, 0, 0);
#else
  *(v8bf*)l = *(const v8bf*)g;
#endif
}

__device__ inline void cp_commit() {
#ifdef ASYNC_LDS
  __builtin_amdgcn_s_wait_asynccnt(0);
#endif
}

// ---------------- fp32 -> bf16 convert (vectorized x4) ----------------
__global__ void f32_to_bf16_vec(const float* __restrict__ in,
                                bf16_t* __restrict__ out, int n4) {
  int i = blockIdx.x * blockDim.x + threadIdx.x;
  if (i < n4) {
    v4f v = ((const v4f*)in)[i];
    v4bf o;
#pragma unroll
    for (int j = 0; j < 4; ++j) o[j] = (bf16_t)v[j];
    ((v4bf*)out)[i] = o;
  }
}

// ---------------- LDS fragment gathers (contiguous-k layouts) ----------------
// A tile in LDS: [rows][32 bf16], stride 32. A-frag: lane l -> row l%16,
// lanes 0-15 hold K 0..15, lanes 16-31 hold K 16..31 (contiguous).
__device__ inline v16bf lds_a_frag(const bf16_t* tileA, int rowBase, int lane) {
  int r  = rowBase + (lane & 15);
  int hf = (lane >> 4) & 1;
  const bf16_t* p = tileA + r * 32;
  v8bf lo = *(const v8bf*)(p + hf * 16);
  v8bf hi = *(const v8bf*)(p + hf * 16 + 8);
  v16bf a;
#pragma unroll
  for (int i = 0; i < 8; ++i) { a[i] = lo[i]; a[i + 8] = hi[i]; }
  return a;
}

// B tile in LDS: [N][32 bf16] (K contiguous per N row). B-frag: lane l holds
// N = nBase + l%16, K = 0..15 (lanes 0-15) or 16..31 (lanes 16-31), contiguous.
__device__ inline v16bf lds_b_frag(const bf16_t* tileB, int nBase, int lane) {
  int n  = nBase + (lane & 15);
  int hf = (lane >> 4) & 1;
  return *(const v16bf*)(tileB + n * 32 + hf * 16);
}

// ---------------- QKV projection GEMM + fused RoPE ----------------
// M=8192 (b*s), N=3072 (3*H*dk), K=1024. Block tile 128x64, 8 waves.
// Double-buffered async LDS staging.
__global__ __launch_bounds__(256) void qkv_rope_gemm(
    const bf16_t* __restrict__ X,   // [8192][1024]
    const bf16_t* __restrict__ W,   // [3072][1024]
    bf16_t* __restrict__ Q, bf16_t* __restrict__ K, bf16_t* __restrict__ V) {
  __shared__ __align__(32) bf16_t tA[2][128 * 32];
  __shared__ __align__(32) bf16_t tB[2][64 * 32];
  const int mBase = blockIdx.x * 128;
  const int nBase = blockIdx.y * 64;
  const int t = threadIdx.x, lane = t & 31, w = t >> 5;

  v8f acc[4];
#pragma unroll
  for (int nt = 0; nt < 4; ++nt)
#pragma unroll
    for (int i = 0; i < 8; ++i) acc[nt][i] = 0.f;

  const int rA = t >> 1, cA = (t & 1) * 16;
  const int rB = t >> 2, cB = (t & 3) * 8;
  const bf16_t* gA = X + (mBase + rA) * 1024 + cA;
  const bf16_t* gB = W + (nBase + rB) * 1024 + cB;

  auto stage = [&](int buf, int k0) {
    cp16(&tA[buf][rA * 32 + cA],     gA + k0);
    cp16(&tA[buf][rA * 32 + cA + 8], gA + k0 + 8);
    cp16(&tB[buf][rB * 32 + cB],     gB + k0);
  };

  stage(0, 0);
  cp_commit();
  __syncthreads();

  for (int k0 = 0; k0 < 1024; k0 += 32) {
    const int cur = (k0 >> 5) & 1;
    if (k0 + 32 < 1024) stage(cur ^ 1, k0 + 32);

    v16bf a  = lds_a_frag(tA[cur], w * 16, lane);
    v16bf b0 = lds_b_frag(tB[cur], 0,  lane);
    v16bf b1 = lds_b_frag(tB[cur], 16, lane);
    v16bf b2 = lds_b_frag(tB[cur], 32, lane);
    v16bf b3 = lds_b_frag(tB[cur], 48, lane);
    acc[0] = __builtin_amdgcn_wmma_f32_16x16x32_bf16(false, a, false, b0,
                                                     (short)0, acc[0], false, false);
    acc[1] = __builtin_amdgcn_wmma_f32_16x16x32_bf16(false, a, false, b1,
                                                     (short)0, acc[1], false, false);
    acc[2] = __builtin_amdgcn_wmma_f32_16x16x32_bf16(false, a, false, b2,
                                                     (short)0, acc[2], false, false);
    acc[3] = __builtin_amdgcn_wmma_f32_16x16x32_bf16(false, a, false, b3,
                                                     (short)0, acc[3], false, false);
    cp_commit();
    __syncthreads();
  }

  // Epilogue: RoPE on q,k (which<2), scatter to Q/K/V [B,H,S,dk] bf16.
  const int hf = lane >> 4, col = lane & 15;
#pragma unroll
  for (int nt = 0; nt < 4; ++nt) {
    const int e     = nBase + nt * 16 + col;   // uniform "which" per fragment
    const int which = e >> 10;
    const int rem   = e & 1023;
    const int h     = rem >> 6;
    const int idx   = rem & 63;
#pragma unroll
    for (int i = 0; i < 8; ++i) {
      const int m = mBase + w * 16 + i + hf * 8;
      const int s = m & (S_LEN - 1);
      float v = acc[nt][i];
      float outv = v;
      if (which < 2) {  // wave-uniform branch
        float partner = __shfl_xor(v, 1, 32);
        int p = idx >> 1;
        float inv = __expf(-0.28782313662425572f * (float)p);  // 10000^(-2p/64)
        float ang = (float)s * inv;
        float c = __cosf(ang), sn = __sinf(ang);
        outv = (idx & 1) ? (partner * sn + v * c) : (v * c - partner * sn);
      }
      const int b = m >> 11;
      const long base = ((long)(b * NHEAD + h) * S_LEN + s) * DK + idx;
      bf16_t ov = (bf16_t)outv;
      if (which == 0)      Q[base] = ov;
      else if (which == 1) K[base] = ov;
      else                 V[base] = ov;
    }
  }
}

// ---------------- flash attention (causal, online softmax) ----------------
// grid: (S/128, B*H). Block = 8 waves; wave w owns 16 query rows.
__global__ __launch_bounds__(256) void flash_attn(
    const bf16_t* __restrict__ Q, const bf16_t* __restrict__ Kg,
    const bf16_t* __restrict__ Vg, bf16_t* __restrict__ O) {
  __shared__ __align__(32) bf16_t Vt[64 * 32];        // V^T tile [d][key]
  __shared__ __align__(32) bf16_t Pst[8][16 * 32];    // per-wave P staging

  const int t = threadIdx.x, lane = t & 31, w = t >> 5;
  const int hf = lane >> 4, col = lane & 15;
  const int q0 = blockIdx.x * 128;
  const int bh = blockIdx.y;
  const bf16_t* qp = Q  + (long)bh * S_LEN * DK;
  const bf16_t* kp = Kg + (long)bh * S_LEN * DK;
  const bf16_t* vp = Vg + (long)bh * S_LEN * DK;
  const int qRow0 = q0 + w * 16;

  // Q A-fragments for d-blocks 0..31 and 32..63 (straight from global).
  v16bf qf[2];
#pragma unroll
  for (int f = 0; f < 2; ++f) {
    const bf16_t* p = qp + (qRow0 + col) * DK + f * 32 + hf * 16;
    v8bf lo = *(const v8bf*)p;
    v8bf hi = *(const v8bf*)(p + 8);
#pragma unroll
    for (int i = 0; i < 8; ++i) { qf[f][i] = lo[i]; qf[f][i + 8] = hi[i]; }
  }

  v8f oacc[4];
  float mrow[8], lrow[8];
#pragma unroll
  for (int i = 0; i < 8; ++i) { mrow[i] = -1e30f; lrow[i] = 0.f; }
#pragma unroll
  for (int nd = 0; nd < 4; ++nd)
#pragma unroll
    for (int i = 0; i < 8; ++i) oacc[nd][i] = 0.f;

  const int kEnd = q0 + 128;
  for (int kBase = 0; kBase < kEnd; kBase += 32) {
    // Stage V^T [64 d][32 key] cooperatively (whole block).
    {
      int key = t & 31, dseg = (t >> 5) * 8;
      v8bf vv = *(const v8bf*)(vp + (kBase + key) * DK + dseg);
#pragma unroll
      for (int j = 0; j < 8; ++j) Vt[(dseg + j) * 32 + key] = vv[j];
    }
    __syncthreads();

    const bool active = (kBase <= qRow0 + 15);  // wave-uniform
    if (active) {
      // scores: Q(16x64) . K^T(64x32); hoist all K B-frags, chain 4 WMMAs
      v8f sc[2];
#pragma unroll
      for (int n = 0; n < 2; ++n)
#pragma unroll
        for (int i = 0; i < 8; ++i) sc[n][i] = 0.f;
      const bf16_t* kb = kp + kBase * DK + hf * 16;
      v16bf kf00 = *(const v16bf*)(kb + col * DK);
      v16bf kf01 = *(const v16bf*)(kb + col * DK + 32);
      v16bf kf10 = *(const v16bf*)(kb + (16 + col) * DK);
      v16bf kf11 = *(const v16bf*)(kb + (16 + col) * DK + 32);
      sc[0] = __builtin_amdgcn_wmma_f32_16x16x32_bf16(false, qf[0], false, kf00,
                                                      (short)0, sc[0], false, false);
      sc[1] = __builtin_amdgcn_wmma_f32_16x16x32_bf16(false, qf[0], false, kf10,
                                                      (short)0, sc[1], false, false);
      sc[0] = __builtin_amdgcn_wmma_f32_16x16x32_bf16(false, qf[1], false, kf01,
                                                      (short)0, sc[0], false, false);
      sc[1] = __builtin_amdgcn_wmma_f32_16x16x32_bf16(false, qf[1], false, kf11,
                                                      (short)0, sc[1], false, false);

      // causal mask + scale + online softmax (rows live in 16-lane halves)
#pragma unroll
      for (int i = 0; i < 8; ++i) {
        const int qrow = qRow0 + i + hf * 8;
#pragma unroll
        for (int n = 0; n < 2; ++n) {
          const int key = kBase + n * 16 + col;
          float s = sc[n][i] * 0.125f;  // 1/sqrt(64)
          sc[n][i] = (key <= qrow) ? s : -1e30f;
        }
        float mx = fmaxf(sc[0][i], sc[1][i]);
        mx = fmaxf(mx, __shfl_xor(mx, 1, 16));
        mx = fmaxf(mx, __shfl_xor(mx, 2, 16));
        mx = fmaxf(mx, __shfl_xor(mx, 4, 16));
        mx = fmaxf(mx, __shfl_xor(mx, 8, 16));
        const float mnew = fmaxf(mrow[i], mx);
        const float rescale = __expf(mrow[i] - mnew);
        lrow[i] *= rescale;
#pragma unroll
        for (int nd = 0; nd < 4; ++nd) oacc[nd][i] *= rescale;
        float p0 = (sc[0][i] > -1e29f) ? __expf(sc[0][i] - mnew) : 0.f;
        float p1 = (sc[1][i] > -1e29f) ? __expf(sc[1][i] - mnew) : 0.f;
        float rs = p0 + p1;
        rs += __shfl_xor(rs, 1, 16);
        rs += __shfl_xor(rs, 2, 16);
        rs += __shfl_xor(rs, 4, 16);
        rs += __shfl_xor(rs, 8, 16);
        lrow[i] += rs;
        mrow[i] = mnew;
        sc[0][i] = p0;
        sc[1][i] = p1;
      }
      // C-layout -> A-layout transpose of P through wave-private LDS
      bf16_t* pw = &Pst[w][0];
#pragma unroll
      for (int n = 0; n < 2; ++n)
#pragma unroll
        for (int i = 0; i < 8; ++i)
          pw[(i + hf * 8) * 32 + n * 16 + col] = (bf16_t)sc[n][i];
      v16bf pa;
      {
        const bf16_t* p = pw + col * 32 + hf * 16;
        v8bf lo = *(const v8bf*)p;
        v8bf hi = *(const v8bf*)(p + 8);
#pragma unroll
        for (int i = 0; i < 8; ++i) { pa[i] = lo[i]; pa[i + 8] = hi[i]; }
      }
      // P(16x32) . V(32x64): hoist 4 Vt B-frags, chain 4 WMMAs
      v16bf vf0 = *(const v16bf*)(&Vt[(col)      * 32 + hf * 16]);
      v16bf vf1 = *(const v16bf*)(&Vt[(16 + col) * 32 + hf * 16]);
      v16bf vf2 = *(const v16bf*)(&Vt[(32 + col) * 32 + hf * 16]);
      v16bf vf3 = *(const v16bf*)(&Vt[(48 + col) * 32 + hf * 16]);
      oacc[0] = __builtin_amdgcn_wmma_f32_16x16x32_bf16(false, pa, false, vf0,
                                                        (short)0, oacc[0], false, false);
      oacc[1] = __builtin_amdgcn_wmma_f32_16x16x32_bf16(false, pa, false, vf1,
                                                        (short)0, oacc[1], false, false);
      oacc[2] = __builtin_amdgcn_wmma_f32_16x16x32_bf16(false, pa, false, vf2,
                                                        (short)0, oacc[2], false, false);
      oacc[3] = __builtin_amdgcn_wmma_f32_16x16x32_bf16(false, pa, false, vf3,
                                                        (short)0, oacc[3], false, false);
    }
    __syncthreads();
  }

  // Write normalized output as bf16 into [B,S,H*dk]
  const int b = bh >> 4, h = bh & 15;
#pragma unroll
  for (int nd = 0; nd < 4; ++nd)
#pragma unroll
    for (int i = 0; i < 8; ++i) {
      const int qrow = qRow0 + i + hf * 8;
      float val = oacc[nd][i] / lrow[i];
      O[(long)(b * S_LEN + qrow) * DMODEL + h * DK + nd * 16 + col] =
          (bf16_t)val;
    }
}

// ---------------- output projection GEMM ----------------
// M=8192, N=1024, K=1024; A = attn bf16, B = W_o bf16, out fp32.
__global__ __launch_bounds__(256) void out_proj_gemm(
    const bf16_t* __restrict__ A, const bf16_t* __restrict__ W,
    float* __restrict__ out) {
  __shared__ __align__(32) bf16_t tA[2][128 * 32];
  __shared__ __align__(32) bf16_t tB[2][64 * 32];
  const int mBase = blockIdx.x * 128;
  const int nBase = blockIdx.y * 64;
  const int t = threadIdx.x, lane = t & 31, w = t >> 5;

  v8f acc[4];
#pragma unroll
  for (int nt = 0; nt < 4; ++nt)
#pragma unroll
    for (int i = 0; i < 8; ++i) acc[nt][i] = 0.f;

  const int rA = t >> 1, cA = (t & 1) * 16;
  const int rB = t >> 2, cB = (t & 3) * 8;
  const bf16_t* gA = A + (mBase + rA) * 1024 + cA;
  const bf16_t* gB = W + (nBase + rB) * 1024 + cB;

  auto stage = [&](int buf, int k0) {
    cp16(&tA[buf][rA * 32 + cA],     gA + k0);
    cp16(&tA[buf][rA * 32 + cA + 8], gA + k0 + 8);
    cp16(&tB[buf][rB * 32 + cB],     gB + k0);
  };

  stage(0, 0);
  cp_commit();
  __syncthreads();

  for (int k0 = 0; k0 < 1024; k0 += 32) {
    const int cur = (k0 >> 5) & 1;
    if (k0 + 32 < 1024) stage(cur ^ 1, k0 + 32);

    v16bf a  = lds_a_frag(tA[cur], w * 16, lane);
    v16bf b0 = lds_b_frag(tB[cur], 0,  lane);
    v16bf b1 = lds_b_frag(tB[cur], 16, lane);
    v16bf b2 = lds_b_frag(tB[cur], 32, lane);
    v16bf b3 = lds_b_frag(tB[cur], 48, lane);
    acc[0] = __builtin_amdgcn_wmma_f32_16x16x32_bf16(false, a, false, b0,
                                                     (short)0, acc[0], false, false);
    acc[1] = __builtin_amdgcn_wmma_f32_16x16x32_bf16(false, a, false, b1,
                                                     (short)0, acc[1], false, false);
    acc[2] = __builtin_amdgcn_wmma_f32_16x16x32_bf16(false, a, false, b2,
                                                     (short)0, acc[2], false, false);
    acc[3] = __builtin_amdgcn_wmma_f32_16x16x32_bf16(false, a, false, b3,
                                                     (short)0, acc[3], false, false);
    cp_commit();
    __syncthreads();
  }

  const int hf = lane >> 4, col = lane & 15;
#pragma unroll
  for (int nt = 0; nt < 4; ++nt)
#pragma unroll
    for (int i = 0; i < 8; ++i) {
      const int m = mBase + w * 16 + i + hf * 8;
      out[(long)m * 1024 + nBase + nt * 16 + col] = acc[nt][i];
    }
}

// ---------------- launcher ----------------
extern "C" void kernel_launch(void* const* d_in, const int* in_sizes, int n_in,
                              void* d_out, int out_size, void* d_ws,
                              size_t ws_size, hipStream_t stream) {
  const float* x    = (const float*)d_in[0];   // 4*2048*1024
  const float* wqkv = (const float*)d_in[1];   // 3072*1024
  const float* wo   = (const float*)d_in[2];   // 1024*1024
  float* out = (float*)d_out;

  char* ws = (char*)d_ws;
  const size_t MB = 1u << 20;
  bf16_t* xh    = (bf16_t*)(ws + 0 * MB);
  bf16_t* wqkvh = (bf16_t*)(ws + 16 * MB);
  bf16_t* woh   = (bf16_t*)(ws + 22 * MB);
  bf16_t* Qb    = (bf16_t*)(ws + 24 * MB);
  bf16_t* Kb    = (bf16_t*)(ws + 40 * MB);
  bf16_t* Vb    = (bf16_t*)(ws + 56 * MB);
  bf16_t* Ab    = (bf16_t*)(ws + 72 * MB);

  const int nX = 4 * S_LEN * DMODEL;       // 8388608
  const int nWqkv = 3 * DMODEL * DMODEL;   // 3145728
  const int nWo = DMODEL * DMODEL;         // 1048576

  f32_to_bf16_vec<<<nX / 4 / 256, 256, 0, stream>>>(x, xh, nX / 4);
  f32_to_bf16_vec<<<nWqkv / 4 / 256, 256, 0, stream>>>(wqkv, wqkvh, nWqkv / 4);
  f32_to_bf16_vec<<<nWo / 4 / 256, 256, 0, stream>>>(wo, woh, nWo / 4);

  qkv_rope_gemm<<<dim3(8192 / 128, 3072 / 64), 256, 0, stream>>>(
      xh, wqkvh, Qb, Kb, Vb);

  flash_attn<<<dim3(S_LEN / 128, 4 * NHEAD), 256, 0, stream>>>(Qb, Kb, Vb, Ab);

  out_proj_gemm<<<dim3(8192 / 128, 1024 / 64), 256, 0, stream>>>(Ab, woh, out);
}
